// Bn_bin_conv_pool_48309792145690
// MI455X (gfx1250) — 8-bit matrix operand order repaired and checked by reading, compiled, not run on hardware
//
#include <hip/hip_runtime.h>
#include <hip/hip_bf16.h>

typedef __attribute__((ext_vector_type(8))) int v8i;

union U32x8 { uint4 q[2]; v8i v; };

#define LCOLS 136
#define LSTRIDE 80

__global__ void pack_w_kernel(const float* __restrict__ W,
                              unsigned int* __restrict__ Apack) {
    int gid = blockIdx.x * blockDim.x + threadIdx.x;
    if (gid >= 8 * 7 * 32 * 8) return;
    int v    = gid & 7;
    int lane = (gid >> 3) & 31;
    int tk   = gid >> 8;
    int k    = tk % 7;
    int t    = tk / 7;
    int m    = lane & 15;
    int half = lane >> 4;
    int kb   = 32 * (v >> 2) + 4 * (v & 3) + 16 * half;
    unsigned int pack = 0;
    for (int b = 0; b < 4; ++b) {
        int ci = kb + b;
        float wv = W[((t * 16 + m) * 64 + ci) * 7 + k];
        int s = (wv > 0.0f) - (wv < 0.0f);
        pack |= ((unsigned int)(unsigned char)(signed char)s) << (8 * b);
    }
    Apack[gid] = pack;
}

__global__ void __launch_bounds__(256)
bconv_kernel(const float* __restrict__ x,
             const float* __restrict__ prelu,
             const unsigned int* __restrict__ Apack,
             float* __restrict__ yout) {
    __shared__ __align__(16) unsigned char sa[LCOLS * LSTRIDE];

    int bid   = blockIdx.x;
    int batch = bid >> 5;
    int lb    = bid & 31;
    int l0    = lb * 128;
    int tid   = threadIdx.x;

    {
        int ci = tid >> 2;
        int j  = tid & 3;
        const float* xbase = x + ((size_t)batch * 64 + ci) * 4096;
        for (int col = j; col < 134; col += 4) {
            int lp = l0 + col - 3;
            float v = -1.0f;
            if (lp >= 0 && lp < 4096) v = xbase[lp];
            signed char s = (signed char)((v > 0.0f) - (v < 0.0f));
            sa[col * LSTRIDE + ci] = (unsigned char)s;
        }
    }

    int w    = tid >> 5;
    int lane = tid & 31;
    v8i a[7];
#pragma unroll
    for (int k = 0; k < 7; ++k) {
        const uint4* src = (const uint4*)(Apack + (((w * 7 + k) * 32 + lane) << 3));
        U32x8 u;
        u.q[0] = src[0];
        u.q[1] = src[1];
        a[k] = u.v;
    }
    __syncthreads();

    float aP   = prelu[0];
    int   nI   = lane & 15;
    int   half = lane >> 4;
    int   co0  = w * 16;
    const unsigned char* bbase = sa + (nI * LSTRIDE + half * 16);

#pragma unroll
    for (int nt = 0; nt < 8; ++nt) {
        U32x8 b[7];
#pragma unroll
        for (int k = 0; k < 7; ++k) {
            const unsigned char* p = bbase + (nt * 16 + k) * LSTRIDE;
            b[k].q[0] = *(const uint4*)(p);
            b[k].q[1] = *(const uint4*)(p + 32);
        }
        v8i acc = {0, 0, 0, 0, 0, 0, 0, 0};
#pragma unroll
        for (int k = 0; k < 7; ++k) {
            acc = __builtin_amdgcn_wmma_i32_16x16x64_iu8(
                 true, a[k],  true, b[k].v, acc,
                 false,  false);
        }
#pragma unroll
        for (int j = 0; j < 8; ++j) {
            int y0 = acc[j];
            int y1 = __shfl_xor(y0, 1, 32);
            int pm = y0 > y1 ? y0 : y1;
            float pf = (float)pm;
            pf = pf >= 0.0f ? pf : aP * pf;
            if ((nI & 1) == 0) {
                int co = co0 + half * 8 + j;
                int Lp = (l0 + nt * 16 + nI) >> 1;
                yout[((size_t)batch * 128 + co) * 2048 + Lp] = pf;
            }
        }
    }
}

__global__ void __launch_bounds__(256)
stats_kernel(const float* __restrict__ y,
             const float* __restrict__ gamma,
             const float* __restrict__ beta,
             float* __restrict__ scale,
             float* __restrict__ shift) {
    int c   = blockIdx.x;
    int tid = threadIdx.x;
    float s = 0.0f, sq = 0.0f;
    for (int n = 0; n < 128; ++n) {
        const float* row = y + ((size_t)(n * 128 + c)) * 2048;
        for (int l = tid; l < 2048; l += 256) {
            float v = row[l];
            s  += v;
            sq += v * v;
        }
    }
    __shared__ float ss[256];
    __shared__ float sqq[256];
    ss[tid] = s;
    sqq[tid] = sq;
    __syncthreads();
    for (int o = 128; o > 0; o >>= 1) {
        if (tid < o) { ss[tid] += ss[tid + o]; sqq[tid] += sqq[tid + o]; }
        __syncthreads();
    }
    if (tid == 0) {
        const float cnt  = 128.0f * 2048.0f;
        float mean = ss[0] / cnt;
        float var  = sqq[0] / cnt - mean * mean;
        float sc   = gamma[c] * rsqrtf(var + 1e-5f);
        scale[c] = sc;
        shift[c] = beta[c] - mean * sc;
    }
}

__global__ void __launch_bounds__(256)
bn_apply_kernel(float* __restrict__ y,
                const float* __restrict__ scale,
                const float* __restrict__ shift) {
    size_t i  = (size_t)blockIdx.x * blockDim.x + threadIdx.x;
    size_t n4 = (size_t)128 * 128 * 2048 / 4;
    if (i >= n4) return;
    int c = (int)((i * 4) / 2048) & 127;
    float4* p = (float4*)y;
    float4 v = p[i];
    float sc = scale[c], sh = shift[c];
    v.x = v.x * sc + sh;
    v.y = v.y * sc + sh;
    v.z = v.z * sc + sh;
    v.w = v.w * sc + sh;
    p[i] = v;
}

extern "C" void kernel_launch(void* const* d_in, const int* in_sizes, int n_in,
                              void* d_out, int out_size, void* d_ws, size_t ws_size,
                              hipStream_t stream) {
    (void)in_sizes; (void)n_in; (void)out_size; (void)ws_size;
    const float* x     = (const float*)d_in[0];
    const float* W     = (const float*)d_in[1];
    const float* prelu = (const float*)d_in[2];
    const float* gamma = (const float*)d_in[3];
    const float* beta  = (const float*)d_in[4];
    float* out = (float*)d_out;

    unsigned int* Apack = (unsigned int*)d_ws;
    float* scale = (float*)((char*)d_ws + 14336 * 4);
    float* shift = scale + 128;

    pack_w_kernel<<<56, 256, 0, stream>>>(W, Apack);
    bconv_kernel<<<128 * 32, 256, 0, stream>>>(x, prelu, Apack, out);
    stats_kernel<<<128, 256, 0, stream>>>(out, gamma, beta, scale, shift);
    bn_apply_kernel<<<(int)(((size_t)128 * 128 * 2048 / 4 + 255) / 256), 256, 0, stream>>>(out, scale, shift);
}
